// IDEncoder_42915313221989
// MI455X (gfx1250) — matrix operand order repaired and checked by reading, not compiled, not run on hardware
//
#include <hip/hip_runtime.h>

typedef _Float16 v4h  __attribute__((ext_vector_type(4)));
typedef _Float16 v8h  __attribute__((ext_vector_type(8)));
typedef _Float16 v16h __attribute__((ext_vector_type(16)));
typedef float    v8f  __attribute__((ext_vector_type(8)));
typedef float    f32x4 __attribute__((ext_vector_type(4)));
typedef unsigned int u32x4 __attribute__((ext_vector_type(4)));
typedef int      i32x4 __attribute__((ext_vector_type(4)));
typedef int      i32x8 __attribute__((ext_vector_type(8)));

#define TA_STR 72     // [64][64+8] f16   A-operand GEMM1
#define W1T_STR 72    // [256][64+8] f16  B-operand GEMM1 (transposed)
#define W2T_STR 264   // [64][256+8] f16  B-operand GEMM2 (transposed)
#define HA_STR 264    // [64][256+8] f16  A-operand GEMM2

// LDS byte offsets (all 16B aligned). tF32 (16KB, TDM landing zone) overlays hA:
// hA is first written in the epilogue, one barrier after the f32->f16 convert.
#define OFF_HA    0                         // 64*264*2 = 33792  (tF32 uses [0,16384))
#define OFF_W2T   (OFF_HA  + 33792)         // 64*264*2 = 33792
#define OFF_W1T   (OFF_W2T + 33792)         // 256*72*2 = 36864
#define OFF_W1G   (OFF_W1T + 36864)         // 64*256*2 = 32768
#define OFF_TA    (OFF_W1G + 32768)         // 64*72*2  =  9216
#define OFF_B1    (OFF_TA  + 9216)          // 1024
#define OFF_B2    (OFF_B1  + 1024)          // 256
#define OFF_XS    (OFF_B2  + 256)           // 256
#define OFF_M     (OFF_XS  + 256)           // 256
#define OFF_RANK  (OFF_M   + 256)           // 256
#define LDS_BYTES (OFF_RANK + 256)          // 148480

__device__ __forceinline__ float fast_tanh(float u) {
#if __has_builtin(__builtin_amdgcn_tanhf)
    return __builtin_amdgcn_tanhf(u);          // v_tanh_f32 (TRANS32, co-issues)
#else
    // tanh(u) = 1 - 2/(e^{2u}+1); exact limits at +/-inf, no NaN (denom >= 1)
    float e = __expf(2.0f * u);                // v_mul + v_exp_f32
    return 1.0f - 2.0f * __builtin_amdgcn_rcpf(e + 1.0f);
#endif
}

__device__ __forceinline__ float gelu_tanh(float x) {
    float x3 = x * x * x;
    float u  = 0.7978845608028654f * (x + 0.044715f * x3);
    float t  = fast_tanh(u);
    float hx = 0.5f * x;
    return hx + hx * t;                        // 0.5*x*(1+tanh(u))
}

__device__ __forceinline__ v16h cat16(v8h lo, v8h hi) {
    return __builtin_shufflevector(lo, hi, 0,1,2,3,4,5,6,7,8,9,10,11,12,13,14,15);
}

__global__ void __launch_bounds__(256, 2)
idenc_kernel(const float* __restrict__ xp, const float* __restrict__ tp,
             const float* __restrict__ W1p, const float* __restrict__ b1p,
             const float* __restrict__ W2p, const float* __restrict__ b2p,
             const unsigned char* __restrict__ maskp, float* __restrict__ outp) {
    extern __shared__ __align__(16) char smem[];
    _Float16* hA   = (_Float16*)(smem + OFF_HA);
    float*    tF32 = (float*)(smem + OFF_HA);     // overlay (dead before epilogue)
    _Float16* sW2T = (_Float16*)(smem + OFF_W2T);
    _Float16* sW1T = (_Float16*)(smem + OFF_W1T);
    _Float16* sW1g = (_Float16*)(smem + OFF_W1G);
    _Float16* tA   = (_Float16*)(smem + OFF_TA);
    float*    sB1  = (float*)(smem + OFF_B1);
    float*    sB2  = (float*)(smem + OFF_B2);
    float*    sXs  = (float*)(smem + OFF_XS);
    float*    sM   = (float*)(smem + OFF_M);
    int*      sRank= (int*)(smem + OFF_RANK);

    const int b   = blockIdx.x;
    const int tid = threadIdx.x;
    const int w   = tid >> 5;
    const int lane= tid & 31;
    const int hs  = (lane >> 4) & 1;
    const int ln  = lane & 15;
    const int mt  = w >> 1;

#if __has_builtin(__builtin_amdgcn_tensor_load_to_lds)
    // ---- TDM: DMA the 64x64 f32 t-tile into LDS (wave 0 issues once) ----
    if (w == 0) {
        unsigned long long ga = (unsigned long long)(tp + (size_t)b * 4096);
        u32x4 g0; i32x8 g1; i32x4 g2, g3; i32x8 g4;
        g0[0] = 1u;                                   // count=1, user mode
        g0[1] = (unsigned)OFF_HA;                     // lds_addr (bytes)
        g0[2] = (unsigned)(ga & 0xFFFFFFFFu);         // global_addr[31:0]
        g0[3] = (unsigned)((ga >> 32) & 0x01FFFFFFu) | 0x80000000u; // addr[56:32] | type=2
        g1[0] = 2 << 16;                              // data_size=4B; mask=0; flags=0
        g1[1] = (int)(64u << 16);                     // tensor_dim0[15:0]=64
        g1[2] = (int)(64u << 16);                     // tensor_dim0 hi=0, tensor_dim1[15:0]=64
        g1[3] = (int)(64u << 16);                     // tensor_dim1 hi=0, tile_dim0=64
        g1[4] = 64;                                   // tile_dim1=64, tile_dim2=0
        g1[5] = 64;                                   // tensor_dim0_stride=64
        g1[6] = 0; g1[7] = 0;                         // stride hi / dim1_stride unused
        g2[0] = 1; g2[1] = 0; g2[2] = 0; g2[3] = 0;   // tensor_dim2=1; tile_dim3 unused
        g3[0] = 0; g3[1] = 0; g3[2] = 0; g3[3] = 0;
        g4[0] = 0; g4[1] = 0; g4[2] = 0; g4[3] = 0;
        g4[4] = 0; g4[5] = 0; g4[6] = 0; g4[7] = 0;
        __builtin_amdgcn_tensor_load_to_lds(g0, g1, g2, g3, g4, 0);
    }
#endif

    // ---- Stage weights -> LDS f16 (vectorized, L2-resident after 1st block) ----
    // W1 top (64x256) -> sW1T[c][r], pack 4 along r (4 coalesced b32 + 1 b64 store)
    for (int it = tid; it < 256 * 16; it += 256) {
        int c = it & 255, r0 = (it >> 8) * 4;
        v4h p = { (_Float16)W1p[(r0+0)*256 + c], (_Float16)W1p[(r0+1)*256 + c],
                  (_Float16)W1p[(r0+2)*256 + c], (_Float16)W1p[(r0+3)*256 + c] };
        *(v4h*)(sW1T + c * W1T_STR + r0) = p;
    }
    // W1 bottom (64x256) row-major -> sW1g (b128 load + b64 store)
    for (int it = tid; it < 4096; it += 256) {
        int off = it * 4;
        f32x4 v = *(const f32x4*)(W1p + 64 * 256 + off);
        v4h p = { (_Float16)v[0], (_Float16)v[1], (_Float16)v[2], (_Float16)v[3] };
        *(v4h*)(sW1g + off) = p;
    }
    // W2 (256x64) -> sW2T[n][k], pack 4 along k
    for (int it = tid; it < 64 * 16; it += 256) {
        int n = it & 63, k0 = (it >> 6) * 4;
        v4h p = { (_Float16)W2p[(k0+0)*64 + n], (_Float16)W2p[(k0+1)*64 + n],
                  (_Float16)W2p[(k0+2)*64 + n], (_Float16)W2p[(k0+3)*64 + n] };
        *(v4h*)(sW2T + n * W2T_STR + k0) = p;
    }
#if !__has_builtin(__builtin_amdgcn_tensor_load_to_lds)
    for (int it = tid; it < 1024; it += 256) {      // fallback: direct t load
        int off = it * 4;
        f32x4 v = *(const f32x4*)(tp + (size_t)b * 4096 + off);
        v4h p = { (_Float16)v[0], (_Float16)v[1], (_Float16)v[2], (_Float16)v[3] };
        *(v4h*)(tA + (off >> 6) * TA_STR + (off & 63)) = p;
    }
#endif
    if (tid < 256) sB1[tid] = b1p[tid];
    if (tid < 64)  sB2[tid] = b2p[tid];
    if (tid < 64) {
        unsigned char mk = maskp[(size_t)b * 64 + tid];
        sXs[tid] = mk ? xp[((size_t)b * 64 + tid) * 16] : 0.0f;
        sM[tid]  = mk ? 1.0f : 0.0f;
    }
#if __has_builtin(__builtin_amdgcn_tensor_load_to_lds)
    if (w == 0) __builtin_amdgcn_s_wait_tensorcnt(0);
#endif
    __syncthreads();

#if __has_builtin(__builtin_amdgcn_tensor_load_to_lds)
    // ---- Convert DMA'd t (f32, LDS) -> tA (f16, A-operand layout) ----
    for (int it = tid; it < 1024; it += 256) {
        int off = it * 4;
        f32x4 v = *(const f32x4*)(tF32 + off);
        v4h p = { (_Float16)v[0], (_Float16)v[1], (_Float16)v[2], (_Float16)v[3] };
        *(v4h*)(tA + (off >> 6) * TA_STR + (off & 63)) = p;
    }
#endif
    // ---- Ranks: stable-argsort inverse ----
    if (tid < 64) {
        float v = sXs[tid];
        int cnt = 0;
        for (int j = 0; j < 64; ++j) {
            float u = sXs[j];
            cnt += (int)((u < v) || (u == v && j < tid));
        }
        sRank[tid] = cnt;   // ascending rank; decreasing rank = 63 - cnt
    }
    __syncthreads();

    // ---- GEMM1: C1[64x256] = tA[64x64] @ W1top[64x256], C1 in registers ----
    v16h aF1[2];
    {
        const int arow = mt * 16 + ln;
        #pragma unroll
        for (int kt = 0; kt < 2; ++kt) {
            int abase = arow * TA_STR + kt * 32 + (hs ? 8 : 0);
            v8h lo = *(const v8h*)(tA + abase);
            v8h hi = *(const v8h*)(tA + abase + 16);
            aF1[kt] = cat16(lo, hi);
        }
    }
    v8f c1[8];
    #pragma unroll
    for (int i = 0; i < 8; ++i) {
        int bn = ((w & 1) * 8 + i) * 16 + ln;
        c1[i] = (v8f){0.f,0.f,0.f,0.f,0.f,0.f,0.f,0.f};
        #pragma unroll
        for (int kt = 0; kt < 2; ++kt) {
            int bbase = bn * W1T_STR + kt * 32 + (hs ? 8 : 0);
            v8h lo = *(const v8h*)(sW1T + bbase);
            v8h hi = *(const v8h*)(sW1T + bbase + 16);
            c1[i] = __builtin_amdgcn_wmma_f32_16x16x32_f16(
                false, aF1[kt], false, cat16(lo, hi), (short)0, c1[i], false, false);
        }
    }

    // ---- Two encodings: gather + gelu -> hA, then GEMM2 accumulate ----
    v8f acc[2];
    acc[0] = (v8f){0.f,0.f,0.f,0.f,0.f,0.f,0.f,0.f};
    acc[1] = (v8f){0.f,0.f,0.f,0.f,0.f,0.f,0.f,0.f};

    for (int e = 0; e < 2; ++e) {
        int rk8[8];
        #pragma unroll
        for (int r = 0; r < 8; ++r) {
            int ri = sRank[mt * 16 + r + hs * 8];
            rk8[r] = e ? ri : 63 - ri;
        }
        #pragma unroll
        for (int i = 0; i < 8; ++i) {
            int col = ((w & 1) * 8 + i) * 16 + ln;
            float b1v = sB1[col];
            #pragma unroll
            for (int r = 0; r < 8; ++r) {
                float val = c1[i][r] + (float)sW1g[rk8[r] * 256 + col] + b1v;
                hA[(mt * 16 + r + hs * 8) * HA_STR + col] = (_Float16)gelu_tanh(val);
            }
        }
        __syncthreads();

        const int arow = mt * 16 + ln;
        #pragma unroll
        for (int kt = 0; kt < 8; ++kt) {
            int abase = arow * HA_STR + kt * 32 + (hs ? 8 : 0);
            v8h alo = *(const v8h*)(hA + abase);
            v8h ahi = *(const v8h*)(hA + abase + 16);
            v16h aF = cat16(alo, ahi);
            #pragma unroll
            for (int j = 0; j < 2; ++j) {
                int bbase = (((w & 1) * 2 + j) * 16 + ln) * W2T_STR + kt * 32 + (hs ? 8 : 0);
                v8h blo = *(const v8h*)(sW2T + bbase);
                v8h bhi = *(const v8h*)(sW2T + bbase + 16);
                acc[j] = __builtin_amdgcn_wmma_f32_16x16x32_f16(
                    false, aF, false, cat16(blo, bhi), (short)0, acc[j], false, false);
            }
        }
        if (e == 0) __syncthreads();   // protect hA before rewrite
    }

    // ---- Final: out = m * (0.5*(h1+h2)@W2 + b2) ----
    #pragma unroll
    for (int j = 0; j < 2; ++j) {
        int col = ((w & 1) * 2 + j) * 16 + ln;
        float b2v = sB2[col];
        #pragma unroll
        for (int r = 0; r < 8; ++r) {
            int rowl = mt * 16 + r + hs * 8;
            outp[((size_t)b * 64 + rowl) * 64 + col] = sM[rowl] * (0.5f * acc[j][r] + b2v);
        }
    }
}

extern "C" void kernel_launch(void* const* d_in, const int* in_sizes, int n_in,
                              void* d_out, int out_size, void* d_ws, size_t ws_size,
                              hipStream_t stream) {
    const float* x  = (const float*)d_in[0];
    const float* t  = (const float*)d_in[1];
    const float* W1 = (const float*)d_in[2];
    const float* b1 = (const float*)d_in[3];
    const float* W2 = (const float*)d_in[4];
    const float* b2 = (const float*)d_in[5];
    const unsigned char* mask = (const unsigned char*)d_in[6];
    float* out = (float*)d_out;

    int B = in_sizes[1] / (64 * 64);
    idenc_kernel<<<dim3(B), dim3(256), LDS_BYTES, stream>>>(x, t, W1, b1, W2, b2, mask, out);
}
